// HydraVideo_53738630807947
// MI455X (gfx1250) — matrix operand order repaired and checked by reading, not compiled, not run on hardware
//
#include <hip/hip_runtime.h>

#define D_MODEL   1024
#define D_INNER   2048
#define NHEADS    32
#define HEADDIM   64
#define D_STATE   64
#define GN        64
#define CONV_DIM  2304
#define D_IN_PROJ 4416
#define K_SP      7
#define K_TP      4
#define CHUNKSZ   256
#define BATCHN    2
#define FRAMES    8
#define SPATIAL   256
#define SEQLEN    2048
#define NTOK      (BATCHN*SEQLEN)   /* 4096 */
#define RSEQ      (2*BATCHN)        /* 4 */
#define NCH       (SEQLEN/CHUNKSZ)  /* 8 */

typedef __attribute__((ext_vector_type(16))) __bf16 v16bf;
typedef __attribute__((ext_vector_type(8)))  float  v8f;
typedef __attribute__((ext_vector_type(16))) unsigned short u16x16;
typedef __attribute__((ext_vector_type(4)))  unsigned su4;
typedef __attribute__((ext_vector_type(8)))  unsigned su8;

union FragU { u16x16 u; v16bf b; };

#define WMMA_BF16(a,b,c) __builtin_amdgcn_wmma_f32_16x16x32_bf16(false,(a),false,(b),(short)0,(c),false,false)

__device__ __forceinline__ unsigned short f2bf(float x) {
    unsigned u = __float_as_uint(x);
    unsigned r = u + 0x7FFFu + ((u >> 16) & 1u);
    return (unsigned short)(r >> 16);
}
__device__ __forceinline__ float bf2f(unsigned short h) {
    return __uint_as_float(((unsigned)h) << 16);
}
__device__ __forceinline__ float siluf(float x) { return x / (1.f + __expf(-x)); }

// ---- async memory -> LDS (16B per lane), tracked on ASYNCcnt
__device__ __forceinline__ void async_b128(const void* gsrc, void* ldst) {
    asm volatile("global_load_async_to_lds_b128 %0, %1, off"
                 :: "v"((unsigned)(size_t)ldst),
                    "v"((unsigned long long)(size_t)gsrc)
                 : "memory");
}
__device__ __forceinline__ void wait_async0() {
    asm volatile("s_wait_asynccnt 0x0" ::: "memory");
}

// ---- Tensor Data Mover: load a 256x64 bf16 tile (128B rows, contiguous in
// memory) into LDS with a 144B padded row stride (CSTR=72 elems).
// 2D descriptor: groups 2/3 NULL. pad: every 32 DWORDs insert 4 DWORDs.
__device__ __forceinline__ void tdm_load_tile_256x64_bf16(
    const unsigned short* gsrc, unsigned short* ldst)
{
    unsigned long long ga = (unsigned long long)(size_t)gsrc;
    su4 g0;
    g0[0] = 1u;                                             // count=1, user mode
    g0[1] = (unsigned)(size_t)ldst;                         // lds_addr
    g0[2] = (unsigned)ga;                                   // global_addr[31:0]
    g0[3] = (unsigned)((ga >> 32) & 0x1FFFFFFu) | (2u<<30); // global_addr[56:32] | type=2
    su8 g1;
    g1[0] = (1u<<16) | (1u<<20) | (4u<<22) | (3u<<25);      // data_size=2B, pad_en, int=32dw, amt=4dw
    g1[1] = (64u << 16);                                    // tensor_dim0 = 64
    g1[2] = (256u << 16);                                   // tensor_dim1 = 256
    g1[3] = (64u << 16);                                    // tile_dim0 = 64
    g1[4] = 256u;                                           // tile_dim1 = 256
    g1[5] = 64u;                                            // tensor_dim0_stride = 64
    g1[6] = 0u;
    g1[7] = 0u;
    asm volatile("tensor_load_to_lds %0, %1" :: "s"(g0), "s"(g1) : "memory");
}

// A-operand fragment: 16(M) x 32(K) bf16 from [row][k] layout, ld elements/row.
__device__ __forceinline__ v16bf load_fragA(const unsigned short* src, int ld) {
    unsigned l = threadIdx.x & 31u;
    const unsigned short* row = src + (l & 15u) * ld + ((l >> 4) ? 8 : 0);
    FragU f;
#pragma unroll
    for (int j = 0; j < 8; ++j) { f.u[j] = row[j]; f.u[j + 8] = row[16 + j]; }
    return f.b;
}
// B-operand fragment: 32(K) x 16(N) bf16 from [n][k] (N-major, K contiguous) layout.
__device__ __forceinline__ v16bf load_fragB(const unsigned short* src, int ld) {
    unsigned l = threadIdx.x & 31u;
    const unsigned short* row = src + (l & 15u) * ld + ((l >> 4) ? 8 : 0);
    FragU f;
#pragma unroll
    for (int j = 0; j < 8; ++j) { f.u[j] = row[j]; f.u[j + 8] = row[16 + j]; }
    return f.b;
}

// ---------------------------------------------------------------- cast
__global__ void cast_k(const float* __restrict__ src, unsigned short* __restrict__ dst, int n) {
    int i = blockIdx.x * 256 + threadIdx.x;
    if (i < n) dst[i] = f2bf(src[i]);
}

// ---------------------------------------------------------------- GEMM: C[MxN] = A[MxK](bf16) * B[NxK](bf16)^T
#define BM 128
#define BN 64
#define BK 32
#define BKP 40
__global__ __launch_bounds__(256) void gemm_bf16_nt(
    const unsigned short* __restrict__ A, const unsigned short* __restrict__ B,
    float* __restrict__ C, int M, int N, int K)
{
    __shared__ unsigned short As[BM * BKP];
    __shared__ unsigned short Bs[BN * BKP];
    int bm = blockIdx.y * BM, bn = blockIdx.x * BN;
    int tid = threadIdx.x, wave = tid >> 5;
    unsigned l = tid & 31u;
    v8f acc[4] = {v8f{0}, v8f{0}, v8f{0}, v8f{0}};
    for (int k0 = 0; k0 < K; k0 += BK) {
        {   // A tile 128x32: two async b128 per thread, no VGPR staging
            int r = tid >> 1, cc = (tid & 1) * 16;
            const unsigned short* ga = A + (size_t)(bm + r) * K + k0 + cc;
            unsigned short* la = As + r * BKP + cc;
            __builtin_prefetch(ga + 2 * BK, 0, 1);
            async_b128(ga, la);
            async_b128(ga + 8, la + 8);
        }
        {   // B tile 64x32 (N-major): one async b128 per thread
            int r = tid >> 2, cc = (tid & 3) * 8;
            int n = bn + r;
            unsigned short* lb = Bs + r * BKP + cc;
            if (n < N) {
                async_b128(B + (size_t)n * K + k0 + cc, lb);
            } else {
#pragma unroll
                for (int j = 0; j < 8; ++j) lb[j] = 0;
            }
        }
        wait_async0();
        __syncthreads();
        v16bf a = load_fragA(As + (wave * 16) * BKP, BKP);
#pragma unroll
        for (int nt = 0; nt < 4; ++nt) {
            v16bf b = load_fragB(Bs + (nt * 16) * BKP, BKP);
            acc[nt] = WMMA_BF16(a, b, acc[nt]);
        }
        __syncthreads();
    }
    int col0 = l & 15, rofs = (l >> 4) ? 8 : 0;
    int m0 = bm + wave * 16;
#pragma unroll
    for (int nt = 0; nt < 4; ++nt) {
        int n = bn + nt * 16 + col0;
        if (n < N) {
#pragma unroll
            for (int i = 0; i < 8; ++i)
                C[(size_t)(m0 + rofs + i) * N + n] = acc[nt][i];
        }
    }
}

// ---------------------------------------------------------------- depthwise convs (+SiLU)
__global__ void conv_sp_k(const float* __restrict__ zxb, const float* __restrict__ w,
                          const float* __restrict__ bias, float* __restrict__ out) {
    size_t idx = (size_t)blockIdx.x * 256 + threadIdx.x;
    if (idx >= (size_t)NTOK * CONV_DIM) return;
    int ci = (int)(idx % CONV_DIM);
    size_t tok = idx / CONV_DIM;
    int b = (int)(tok / SEQLEN), lcur = (int)(tok % SEQLEN);
    int s = lcur >> 3, fr = lcur & 7;
    float acc = bias[ci];
#pragma unroll
    for (int k = 0; k < K_SP; ++k) {
        int ss = s + k - 3;
        if (ss >= 0 && ss < SPATIAL)
            acc += w[ci * K_SP + k] *
                   zxb[((size_t)(b * SEQLEN + (ss << 3) + fr)) * D_IN_PROJ + D_INNER + ci];
    }
    out[idx] = siluf(acc);
}
__global__ void conv_tp_k(const float* __restrict__ xin, const float* __restrict__ w,
                          const float* __restrict__ bias, float* __restrict__ out) {
    size_t idx = (size_t)blockIdx.x * 256 + threadIdx.x;
    if (idx >= (size_t)NTOK * CONV_DIM) return;
    int ci = (int)(idx % CONV_DIM);
    size_t tok = idx / CONV_DIM;
    int b = (int)(tok / SEQLEN), lcur = (int)(tok % SEQLEN);
    int s = lcur >> 3, fr = lcur & 7;
    float acc = bias[ci];
#pragma unroll
    for (int k = 0; k < K_TP; ++k) {
        int fr2 = fr + k - 3;
        if (fr2 >= 0)
            acc += w[ci * K_TP + k] *
                   xin[((size_t)(b * SEQLEN + (s << 3) + fr2)) * CONV_DIM + ci];
    }
    out[idx] = siluf(acc);
}

// ---------------------------------------------------------------- prep x2/B/C (bf16) + dt (softplus)
__global__ void prep_k(const float* __restrict__ xbc2, const float* __restrict__ zxb,
                       const float* __restrict__ dt_bias,
                       unsigned short* __restrict__ x2bf, unsigned short* __restrict__ Bbf,
                       unsigned short* __restrict__ Cbf, float* __restrict__ dtG) {
    int tok = blockIdx.x;               // 0..RSEQ*SEQLEN-1
    int r = tok / SEQLEN, lcur = tok % SEQLEN;
    int b = (r < BATCHN) ? r : r - BATCHN;
    int s = lcur >> 3, fr = lcur & 7;
    int lsrc = (r < BATCHN) ? lcur : ((s << 3) + (7 - fr));
    size_t src = (size_t)(b * SEQLEN + lsrc);
    int tid = threadIdx.x;
#pragma unroll
    for (int j = 0; j < 8; ++j) {
        int d = tid + j * 256;
        x2bf[(size_t)tok * D_INNER + d] = f2bf(xbc2[src * CONV_DIM + d]);
    }
    if (tid < 64) {
        int cofs = (r < BATCHN) ? D_INNER : (D_INNER + 2 * GN);
        Bbf[(size_t)tok * GN + tid] = f2bf(xbc2[src * CONV_DIM + cofs + tid]);
    } else if (tid < 128) {
        int n = tid - 64;
        int cofs = (r < BATCHN) ? (D_INNER + GN) : (D_INNER + 3 * GN);
        Cbf[(size_t)tok * GN + n] = f2bf(xbc2[src * CONV_DIM + cofs + n]);
    } else if (tid < 128 + NHEADS) {
        int hh = tid - 128;
        int dcol = D_INNER + CONV_DIM + ((r < BATCHN) ? hh : NHEADS + hh);
        float v = zxb[src * D_IN_PROJ + dcol] + dt_bias[hh];
        dtG[(size_t)tok * NHEADS + hh] = (v > 20.f) ? v : log1pf(__expf(v));
    }
}

// ---------------------------------------------------------------- per-chunk cumsum of dt*A
__global__ void cumsum_k(const float* __restrict__ dtG, const float* __restrict__ A_log,
                         float* __restrict__ cumG, float* __restrict__ cdecay) {
    int id = blockIdx.x * 256 + threadIdx.x;
    if (id >= RSEQ * NCH * NHEADS) return;
    int h = id % NHEADS, rc = id / NHEADS, c = rc % NCH, r = rc / NCH;
    float Ah = -__expf(A_log[h]);
    size_t tb = (size_t)r * SEQLEN + (size_t)c * CHUNKSZ;
    float cv = 0.f;
    for (int t = 0; t < CHUNKSZ; ++t) {
        cv += dtG[(tb + t) * NHEADS + h] * Ah;
        cumG[(size_t)id * CHUNKSZ + t] = cv;
    }
    cdecay[id] = __expf(cv);
}

// ---------------------------------------------------------------- intra-chunk scan (WMMA heavy)
#define CSTR 72
#define XSTR 264
#define MSTR 40
#define SMEM_SCAN (3*256*4 + 2*256*CSTR*2 + 2*64*XSTR*2 + 8*32*MSTR*2)
__global__ __launch_bounds__(256) void scan_intra_k(
    const unsigned short* __restrict__ x2bf, const unsigned short* __restrict__ Bbf,
    const unsigned short* __restrict__ Cbf, const float* __restrict__ dtG,
    const float* __restrict__ cumG, float* __restrict__ ybuf, float* __restrict__ Sck)
{
    extern __shared__ char smem[];
    float* cum_l = (float*)smem;
    float* dts_l = cum_l + 256;
    float* Gs    = dts_l + 256;
    unsigned short* Cs  = (unsigned short*)(Gs + 256);
    unsigned short* Bsn = Cs + 256 * CSTR;
    unsigned short* xT  = Bsn + 256 * CSTR;
    unsigned short* BGt = xT + 64 * XSTR;
    unsigned short* Ms  = BGt + 64 * XSTR;

    int h = blockIdx.x, c = blockIdx.y, r = blockIdx.z;
    int tid = threadIdx.x, wave = tid >> 5;
    unsigned l = tid & 31u;
    int col0 = l & 15, rofs = (l >> 4) ? 8 : 0;
    size_t tokbase = (size_t)r * SEQLEN + (size_t)c * CHUNKSZ;
    int cidx = (r * NCH + c) * NHEADS + h;

    // phase 1: TDM pulls B/C tiles into padded LDS; other waves stage x^T, cum, dt
    if (tid < 32) {
        tdm_load_tile_256x64_bf16(Cbf + tokbase * GN, Cs);
        tdm_load_tile_256x64_bf16(Bbf + tokbase * GN, Bsn);
        __builtin_amdgcn_s_wait_tensorcnt(0);
    }
    {
        const unsigned short* gx = x2bf + (tokbase + tid) * D_INNER + h * HEADDIM;
#pragma unroll 8
        for (int p = 0; p < HEADDIM; ++p) xT[p * XSTR + tid] = gx[p];
        cum_l[tid] = cumG[(size_t)cidx * CHUNKSZ + tid];
        dts_l[tid] = dtG[(tokbase + tid) * NHEADS + h];
    }
    __syncthreads();
    Gs[tid] = __expf(cum_l[255] - cum_l[tid]) * dts_l[tid];
    __syncthreads();
#pragma unroll 8
    for (int n = 0; n < GN; ++n)
        BGt[n * XSTR + tid] = f2bf(bf2f(Bsn[tid * CSTR + n]) * Gs[tid]);
    __syncthreads();

    // main loop: Y_intra for this wave's 32 t-rows
    v8f Y[2][4];
#pragma unroll
    for (int tt = 0; tt < 2; ++tt)
#pragma unroll
        for (int pt = 0; pt < 4; ++pt) Y[tt][pt] = v8f{0};

    unsigned short* Msw = Ms + wave * 32 * MSTR;
    for (int sb = 0; sb < 8; ++sb) {
#pragma unroll
        for (int tt = 0; tt < 2; ++tt) {
#pragma unroll
            for (int ss = 0; ss < 2; ++ss) {
                v8f cb = v8f{0};
#pragma unroll
                for (int kk = 0; kk < 64; kk += 32) {
                    v16bf a = load_fragA(Cs + (wave * 32 + tt * 16) * CSTR + kk, CSTR);
                    v16bf b = load_fragB(Bsn + (sb * 32 + ss * 16) * CSTR + kk, CSTR);
                    cb = WMMA_BF16(a, b, cb);
                }
                int sg = sb * 32 + ss * 16 + col0;
                float cums = cum_l[sg], ds = dts_l[sg];
#pragma unroll
                for (int i = 0; i < 8; ++i) {
                    int tg = wave * 32 + tt * 16 + rofs + i;
                    float v = (sg <= tg) ? cb[i] * __expf(cum_l[tg] - cums) * ds : 0.f;
                    Msw[(tt * 16 + rofs + i) * MSTR + ss * 16 + col0] = f2bf(v);
                }
            }
        }
        asm volatile("s_wait_dscnt 0x0" ::: "memory");   // per-wave LDS staging RAW
#pragma unroll
        for (int tt = 0; tt < 2; ++tt) {
            v16bf a = load_fragA(Msw + tt * 16 * MSTR, MSTR);
#pragma unroll
            for (int pt = 0; pt < 4; ++pt) {
                v16bf b = load_fragB(xT + pt * 16 * XSTR + sb * 32, XSTR);
                Y[tt][pt] = WMMA_BF16(a, b, Y[tt][pt]);
            }
        }
    }
    // store Y_intra
#pragma unroll
    for (int tt = 0; tt < 2; ++tt)
#pragma unroll
        for (int pt = 0; pt < 4; ++pt)
#pragma unroll
            for (int i = 0; i < 8; ++i) {
                int t = wave * 32 + tt * 16 + rofs + i;
                ybuf[(tokbase + t) * D_INNER + h * HEADDIM + pt * 16 + col0] = Y[tt][pt][i];
            }
    // S_chunk[p,n] = sum_s x[s,p]*G[s]*B[s,n]
    float* Sout = Sck + (size_t)cidx * (HEADDIM * D_STATE);
#pragma unroll
    for (int e = 0; e < 2; ++e) {
        int tix = wave * 2 + e, pr = tix >> 2, nc = tix & 3;
        v8f sacc = v8f{0};
#pragma unroll
        for (int kb = 0; kb < 8; ++kb) {
            v16bf a = load_fragA(xT + pr * 16 * XSTR + kb * 32, XSTR);
            v16bf b = load_fragB(BGt + nc * 16 * XSTR + kb * 32, XSTR);
            sacc = WMMA_BF16(a, b, sacc);
        }
#pragma unroll
        for (int i = 0; i < 8; ++i)
            Sout[(pr * 16 + rofs + i) * D_STATE + nc * 16 + col0] = sacc[i];
    }
}

// ---------------------------------------------------------------- sequential inter-chunk state pass
__global__ void chunk_recur_k(const float* __restrict__ Sck, const float* __restrict__ cdecay,
                              float* __restrict__ prevS) {
    int r = blockIdx.x >> 5, h = blockIdx.x & 31;
    int tid = threadIdx.x;
#pragma unroll
    for (int j = 0; j < 16; ++j) {
        int e = tid + j * 256;
        float S = 0.f;
        for (int c = 0; c < NCH; ++c) {
            if (r >= BATCHN && c > 0) S = 0.f;   // reverse direction resets each chunk
            size_t idx = (size_t)((r * NCH + c) * NHEADS + h) * (HEADDIM * D_STATE) + e;
            prevS[idx] = S;
            S = cdecay[(r * NCH + c) * NHEADS + h] * S + Sck[idx];
        }
    }
}

// ---------------------------------------------------------------- Y_inter = exp(cum) * C * prev^T (WMMA)
__global__ __launch_bounds__(256) void y_inter_k(
    const unsigned short* __restrict__ Cbf, const float* __restrict__ prevS,
    const float* __restrict__ cumG, float* __restrict__ ybuf)
{
    __shared__ unsigned short Cs[256 * CSTR];
    __shared__ unsigned short Pb[64 * CSTR];
    __shared__ float cum_l[256];
    int h = blockIdx.x, c = blockIdx.y, r = blockIdx.z;
    int tid = threadIdx.x, wave = tid >> 5;
    unsigned l = tid & 31u;
    int col0 = l & 15, rofs = (l >> 4) ? 8 : 0;
    size_t tokbase = (size_t)r * SEQLEN + (size_t)c * CHUNKSZ;
    int cidx = (r * NCH + c) * NHEADS + h;

    if (tid < 32) {
        tdm_load_tile_256x64_bf16(Cbf + tokbase * GN, Cs);
        __builtin_amdgcn_s_wait_tensorcnt(0);
    }
    cum_l[tid] = cumG[(size_t)cidx * CHUNKSZ + tid];
    const float* Pg = prevS + (size_t)cidx * (HEADDIM * D_STATE);
#pragma unroll
    for (int j = 0; j < 16; ++j) {
        int e = tid * 16 + j;
        Pb[(e >> 6) * CSTR + (e & 63)] = f2bf(Pg[e]);
    }
    __syncthreads();

#pragma unroll
    for (int tt = 0; tt < 2; ++tt) {
#pragma unroll
        for (int pt = 0; pt < 4; ++pt) {
            v8f acc = v8f{0};
#pragma unroll
            for (int kk = 0; kk < 64; kk += 32) {
                v16bf a = load_fragA(Cs + (wave * 32 + tt * 16) * CSTR + kk, CSTR);
                v16bf b = load_fragB(Pb + pt * 16 * CSTR + kk, CSTR);
                acc = WMMA_BF16(a, b, acc);
            }
#pragma unroll
            for (int i = 0; i < 8; ++i) {
                int t = wave * 32 + tt * 16 + rofs + i;
                size_t yi = (tokbase + t) * D_INNER + h * HEADDIM + pt * 16 + col0;
                ybuf[yi] += __expf(cum_l[t]) * acc[i];
            }
        }
    }
}

// ---------------------------------------------------------------- finalize: shifts + residual + RMSnorm + gate
__global__ __launch_bounds__(256) void finalize_k(
    const float* __restrict__ ybuf, const float* __restrict__ xbc2,
    const float* __restrict__ zxb, const float* __restrict__ xfc,
    const float* __restrict__ Dp, const float* __restrict__ norm_w,
    unsigned short* __restrict__ ynorm)
{
    __shared__ float sred[256];
    int tok = blockIdx.x;
    int b = tok / SEQLEN, lcur = tok % SEQLEN;
    int s = lcur >> 3, fr = lcur & 7;
    int tid = threadIdx.x;
    float vals[8], part = 0.f;
#pragma unroll
    for (int j = 0; j < 8; ++j) {
        int d = tid + j * 256;
        int hh = d >> 6;
        float yf = (lcur == 0) ? 0.f
                 : ybuf[((size_t)(b * SEQLEN + lcur - 1)) * D_INNER + d];
        float yr = (s == 0) ? 0.f
                 : ybuf[((size_t)((b + BATCHN) * SEQLEN + ((s - 1) << 3) + (7 - fr))) * D_INNER + d];
        float xo = xbc2[(size_t)tok * CONV_DIM + d];
        float v = yf + yr + xo * (xfc[(size_t)tok * NHEADS + hh] + Dp[hh]);
        vals[j] = v; part += v * v;
    }
    sred[tid] = part; __syncthreads();
    for (int st = 128; st > 0; st >>= 1) { if (tid < st) sred[tid] += sred[tid + st]; __syncthreads(); }
    float rs = rsqrtf(sred[0] / (float)D_INNER + 1e-5f);
#pragma unroll
    for (int j = 0; j < 8; ++j) {
        int d = tid + j * 256;
        float z = zxb[(size_t)tok * D_IN_PROJ + d];
        ynorm[(size_t)tok * D_INNER + d] = f2bf(vals[j] * rs * norm_w[d] * siluf(z));
    }
}

// ---------------------------------------------------------------- host
extern "C" void kernel_launch(void* const* d_in, const int* in_sizes, int n_in,
                              void* d_out, int out_size, void* d_ws, size_t ws_size,
                              hipStream_t stream) {
    (void)in_sizes; (void)n_in; (void)out_size; (void)ws_size;
    const float* u         = (const float*)d_in[0];
    const float* in_proj_w = (const float*)d_in[1];
    const float* conv_sp_w = (const float*)d_in[2];
    const float* conv_sp_b = (const float*)d_in[3];
    const float* conv_tp_w = (const float*)d_in[4];
    const float* conv_tp_b = (const float*)d_in[5];
    const float* dt_bias   = (const float*)d_in[6];
    const float* A_log     = (const float*)d_in[7];
    const float* Dp        = (const float*)d_in[8];
    const float* fc_D_w    = (const float*)d_in[9];
    const float* norm_w    = (const float*)d_in[10];
    const float* out_proj_w= (const float*)d_in[11];
    float* out = (float*)d_out;

    char* w = (char*)d_ws;
    auto alloc = [&](size_t bytes) { char* p = w; w += (bytes + 255) & ~(size_t)255; return p; };
    unsigned short* u_bf    = (unsigned short*)alloc((size_t)NTOK * D_MODEL * 2);
    unsigned short* win_bf  = (unsigned short*)alloc((size_t)D_IN_PROJ * D_MODEL * 2);
    unsigned short* wout_bf = (unsigned short*)alloc((size_t)D_MODEL * D_INNER * 2);
    unsigned short* wfc_bf  = (unsigned short*)alloc((size_t)NHEADS * D_INNER * 2);
    float* zxbcdt = (float*)alloc((size_t)NTOK * D_IN_PROJ * 4);
    float* xbc1   = (float*)alloc((size_t)NTOK * CONV_DIM * 4);
    float* xbc2   = (float*)alloc((size_t)NTOK * CONV_DIM * 4);
    unsigned short* x2bf = (unsigned short*)alloc((size_t)RSEQ * SEQLEN * D_INNER * 2);
    unsigned short* Bbf  = (unsigned short*)alloc((size_t)RSEQ * SEQLEN * GN * 2);
    unsigned short* Cbf  = (unsigned short*)alloc((size_t)RSEQ * SEQLEN * GN * 2);
    float* dtG    = (float*)alloc((size_t)RSEQ * SEQLEN * NHEADS * 4);
    float* cumG   = (float*)alloc((size_t)RSEQ * NCH * NHEADS * CHUNKSZ * 4);
    float* cdecay = (float*)alloc((size_t)RSEQ * NCH * NHEADS * 4);
    float* Sck    = (float*)alloc((size_t)RSEQ * NCH * NHEADS * HEADDIM * D_STATE * 4);
    float* prevS  = (float*)alloc((size_t)RSEQ * NCH * NHEADS * HEADDIM * D_STATE * 4);
    float* ybuf   = (float*)alloc((size_t)RSEQ * SEQLEN * D_INNER * 4);
    float* xfc    = (float*)alloc((size_t)NTOK * NHEADS * 4);
    unsigned short* ynorm = (unsigned short*)alloc((size_t)NTOK * D_INNER * 2);

    int n;
    n = NTOK * D_MODEL;        cast_k<<<(n + 255) / 256, 256, 0, stream>>>(u, u_bf, n);
    n = D_IN_PROJ * D_MODEL;   cast_k<<<(n + 255) / 256, 256, 0, stream>>>(in_proj_w, win_bf, n);
    n = D_MODEL * D_INNER;     cast_k<<<(n + 255) / 256, 256, 0, stream>>>(out_proj_w, wout_bf, n);
    n = NHEADS * D_INNER;      cast_k<<<(n + 255) / 256, 256, 0, stream>>>(fc_D_w, wfc_bf, n);

    gemm_bf16_nt<<<dim3(D_IN_PROJ / BN, NTOK / BM), 256, 0, stream>>>(
        u_bf, win_bf, zxbcdt, NTOK, D_IN_PROJ, D_MODEL);

    int nconv = NTOK * CONV_DIM;
    conv_sp_k<<<(nconv + 255) / 256, 256, 0, stream>>>(zxbcdt, conv_sp_w, conv_sp_b, xbc1);
    conv_tp_k<<<(nconv + 255) / 256, 256, 0, stream>>>(xbc1, conv_tp_w, conv_tp_b, xbc2);

    prep_k<<<RSEQ * SEQLEN, 256, 0, stream>>>(xbc2, zxbcdt, dt_bias, x2bf, Bbf, Cbf, dtG);
    cumsum_k<<<(RSEQ * NCH * NHEADS + 255) / 256, 256, 0, stream>>>(dtG, A_log, cumG, cdecay);

    gemm_bf16_nt<<<dim3(1, NTOK / BM), 256, 0, stream>>>(
        x2bf, wfc_bf, xfc, NTOK, NHEADS, D_INNER);   // forward half of x2bf == x

    scan_intra_k<<<dim3(NHEADS, NCH, RSEQ), 256, SMEM_SCAN, stream>>>(
        x2bf, Bbf, Cbf, dtG, cumG, ybuf, Sck);

    chunk_recur_k<<<RSEQ * NHEADS, 256, 0, stream>>>(Sck, cdecay, prevS);

    y_inter_k<<<dim3(NHEADS, NCH, RSEQ), 256, 0, stream>>>(Cbf, prevS, cumG, ybuf);

    finalize_k<<<NTOK, 256, 0, stream>>>(ybuf, xbc2, zxbcdt, xfc, Dp, norm_w, ynorm);

    gemm_bf16_nt<<<dim3(D_MODEL / BN, NTOK / BM), 256, 0, stream>>>(
        ynorm, wout_bf, out, NTOK, D_MODEL, D_INNER);
}
